// LoFGAN_39694087749923
// MI455X (gfx1250) — matrix operand order repaired and checked by reading, not compiled, not run on hardware
//
#include <hip/hip_runtime.h>
#include <hip/hip_bf16.h>

#define B_    8
#define K_    8
#define C_    128
#define HW_   4096
#define NUM_  2048
#define NREF_ 7

typedef __bf16 v16bf __attribute__((ext_vector_type(16)));
typedef float  v8f   __attribute__((ext_vector_type(8)));

union ABReg { uint4 u[2]; v16bf v; };

static __device__ __forceinline__ unsigned short f2bf(float f) {
  unsigned u = __float_as_uint(f);
  unsigned r = u + 0x7FFFu + ((u >> 16) & 1u);   // round-to-nearest-even
  return (unsigned short)(r >> 16);
}

// ---------------- kernel 1: copy feat -> out ----------------
__global__ void k_copy(const float4* __restrict__ src, float4* __restrict__ dst, int n4) {
  int i = blockIdx.x * blockDim.x + threadIdx.x;
  if (i < n4) dst[i] = src[i];
}

// ---------------- kernel 2: gather + double-normalize selected feat cols -> Qn bf16 ----------------
__global__ void k_qnorm(const float* __restrict__ feat, const int* __restrict__ fidx,
                        unsigned short* __restrict__ Qn) {
  int b = blockIdx.y;
  int q = blockIdx.x * blockDim.x + threadIdx.x;           // 0..NUM_-1
  int fi = fidx[b * NUM_ + q];
  const float* col = feat + (size_t)b * C_ * HW_ + fi;
  float s = 0.f;
#pragma unroll 4
  for (int c = 0; c < C_; ++c) { float x = col[(size_t)c * HW_]; s += x * x; }
  float inv  = 1.f / fmaxf(sqrtf(s), 1e-12f);
  float s2   = s * inv * inv;                               // second l2norm (matches reference)
  float inv2 = 1.f / fmaxf(sqrtf(s2), 1e-12f);
  float sc   = inv * inv2;
  unsigned short* dst = Qn + ((size_t)b * NUM_ + q) * C_;
#pragma unroll 4
  for (int c = 0; c < C_; ++c) dst[c] = f2bf(col[(size_t)c * HW_] * sc);
}

// ---------------- kernel 3: normalize ref columns over C, transpose -> RnT[bn][hw][c] bf16 ----------------
__global__ void k_rnorm(const float* __restrict__ refs, const int* __restrict__ pindex,
                        unsigned short* __restrict__ RnT) {
  int bn = blockIdx.y; int b = bn / NREF_, n = bn % NREF_;
  int idx0 = *pindex;
  int kk = n + (n >= idx0 ? 1 : 0);
  int hw = blockIdx.x * blockDim.x + threadIdx.x;
  const float* col = refs + (((size_t)b * K_ + kk) * C_) * (size_t)HW_ + hw;
  float s = 0.f;
#pragma unroll 4
  for (int c = 0; c < C_; ++c) { float x = col[(size_t)c * HW_]; s += x * x; }
  float inv = 1.f / fmaxf(sqrtf(s), 1e-12f);
  unsigned short* dst = RnT + ((size_t)bn * HW_ + hw) * C_;
#pragma unroll 4
  for (int c = 0; c < C_; ++c) dst[c] = f2bf(col[(size_t)c * HW_] * inv);
}

// ---------------- kernel 4: WMMA GEMM (2048x4096x128 bf16) + running row-argmax ----------------
// Double-buffered: GLOBAL_LOAD_ASYNC_TO_LDS_B128 DMAs the next 32KB ref tile into LDS
// (ASYNCcnt-tracked, no VGPR round trip) while WMMAs consume the current tile.
__global__ __launch_bounds__(256) void k_gemm_argmax(
    const unsigned short* __restrict__ Qn, const unsigned short* __restrict__ RnT,
    int* __restrict__ refIdx) {
  __shared__ unsigned short Bs[2][128][136];   // 136-short pitch: conflict-free b128 LDS reads

  int bn    = blockIdx.y;
  int b     = bn / NREF_;
  int qtile = blockIdx.x;                      // 16 tiles of 128 Q-rows
  int wave  = threadIdx.x >> 5;
  int lane  = threadIdx.x & 31;
  int m     = lane & 15, h = lane >> 4;
  int rowBase = qtile * 128 + wave * 16;

  // A operand: 16x128 bf16 rows of Q, contiguous k order (k = 16h + s) = two adjacent 16B chunks per lane per K-32 block
  ABReg A[4];
  const uint4* qrow = (const uint4*)(Qn + ((size_t)b * NUM_ + rowBase + m) * C_);
#pragma unroll
  for (int kb = 0; kb < 4; ++kb) {
    int K0 = kb * 32 + 16 * h;
    A[kb].u[0] = qrow[K0 >> 3];
    A[kb].u[1] = qrow[(K0 >> 3) + 1];
  }

  float best[8]; int bidx[8];
#pragma unroll
  for (int r = 0; r < 8; ++r) { best[r] = -3.4e38f; bidx[r] = 0; }

  const unsigned short* rbase = RnT + (size_t)bn * HW_ * C_;
  int rowN  = threadIdx.x >> 1;
  int halfN = threadIdx.x & 1;

  // per-thread slice of a chunk: 128 bytes = 8 async b128 DMAs (8 per wave -> ASYNCcnt += 8)
  auto issue_chunk = [&](int nc, int buf) {
    const unsigned short* src = rbase + ((size_t)(nc * 128 + rowN)) * C_ + halfN * 64;
    unsigned ldsa = (unsigned)(size_t)&Bs[buf][rowN][halfN * 64];
#pragma unroll
    for (int t = 0; t < 8; ++t) {
      asm volatile("global_load_async_to_lds_b128 %0, %1, off"
                   :: "v"(ldsa + (unsigned)(t * 16)), "v"(src + t * 8)
                   : "memory");
    }
  };

  const int NCH = HW_ / 128;
  issue_chunk(0, 0);

  for (int nc = 0; nc < NCH; ++nc) {
    int buf = nc & 1;
    if (nc + 1 < NCH) {
      issue_chunk(nc + 1, buf ^ 1);
      asm volatile("s_wait_asynccnt 8" ::: "memory");   // retire current chunk's 8 (in-order)
    } else {
      asm volatile("s_wait_asynccnt 0" ::: "memory");
    }
    __syncthreads();

#pragma unroll 1
    for (int sub = 0; sub < 8; sub += 2) {     // 2 independent accumulator chains hide WMMA->VALU hazard
      int col0 = sub * 16 + m;
      int col1 = col0 + 16;
      v8f acc0 = {}, acc1 = {};
#pragma unroll
      for (int kb = 0; kb < 4; ++kb) {
        ABReg Bv0, Bv1;                         // B uses A's k order: lane=col, 16 contiguous K per lane (k = 16h + s)
        const uint4* bp0 = (const uint4*)(&Bs[buf][col0][kb * 32 + 16 * h]);
        const uint4* bp1 = (const uint4*)(&Bs[buf][col1][kb * 32 + 16 * h]);
        Bv0.u[0] = bp0[0]; Bv0.u[1] = bp0[1];
        Bv1.u[0] = bp1[0]; Bv1.u[1] = bp1[1];
        acc0 = __builtin_amdgcn_wmma_f32_16x16x32_bf16(
            false, A[kb].v, false, Bv0.v, (short)0, acc0, false, false);
        acc1 = __builtin_amdgcn_wmma_f32_16x16x32_bf16(
            false, A[kb].v, false, Bv1.v, (short)0, acc1, false, false);
      }
      int g0 = nc * 128 + col0;
      int g1 = nc * 128 + col1;
#pragma unroll
      for (int r = 0; r < 8; ++r) {            // plain VALU: co-executes with XDL WMMA
        if (acc0[r] > best[r]) { best[r] = acc0[r]; bidx[r] = g0; }
        if (acc1[r] > best[r]) { best[r] = acc1[r]; bidx[r] = g1; }
      }
    }
    __syncthreads();
  }

  // butterfly reduce (val,idx) across the 16 lanes of each half-wave
#pragma unroll
  for (int r = 0; r < 8; ++r) {
    float v = best[r]; int i = bidx[r];
#pragma unroll
    for (int mask = 1; mask < 16; mask <<= 1) {
      float v2 = __shfl_xor(v, mask, 32);
      int   i2 = __shfl_xor(i, mask, 32);
      if (v2 > v || (v2 == v && i2 < i)) { v = v2; i = i2; }
    }
    if (m == 0) {
      int row = rowBase + r + 8 * h;           // D layout: VGPR r -> M=r (lanes 0-15), M=r+8 (16-31)
      refIdx[(size_t)bn * NUM_ + row] = i;
    }
  }
}

// ---------------- kernel 5: gather best ref columns, weighted fuse, scatter into out ----------------
__global__ void k_fuse(const float* __restrict__ feat, const float* __restrict__ refs,
                       const float* __restrict__ sim, const int* __restrict__ fidx,
                       const int* __restrict__ refIdx, const int* __restrict__ pindex,
                       float* __restrict__ out) {
  int bq = blockIdx.x; int b = bq >> 11; int q = bq & (NUM_ - 1);
  int c = threadIdx.x;
  int idx0 = *pindex;
  int fi = fidx[b * NUM_ + q];
  float acc = sim[b * K_ + idx0] * feat[((size_t)b * C_ + c) * HW_ + fi];
#pragma unroll
  for (int n = 0; n < NREF_; ++n) {
    int kk = n + (n >= idx0 ? 1 : 0);
    int ri = refIdx[((size_t)b * NREF_ + n) * NUM_ + q];
    acc += sim[b * K_ + kk] * refs[(((size_t)b * K_ + kk) * C_ + c) * (size_t)HW_ + ri];
  }
  out[((size_t)b * C_ + c) * HW_ + fi] = acc;
}

// ---------------- kernel 6: emit index outputs (as floats) into concatenated tail ----------------
__global__ void k_idxout(const int* __restrict__ fidx, const int* __restrict__ refIdx,
                         float* __restrict__ out) {
  int i = blockIdx.x * blockDim.x + threadIdx.x;
  const int NF = B_ * NUM_;
  const int NR = B_ * NREF_ * NUM_;
  const size_t base = (size_t)B_ * C_ * HW_;
  if (i < NF)            out[base + i] = (float)fidx[i];
  else if (i < NF + NR)  out[base + i] = (float)refIdx[i - NF];
}

extern "C" void kernel_launch(void* const* d_in, const int* in_sizes, int n_in,
                              void* d_out, int out_size, void* d_ws, size_t ws_size,
                              hipStream_t stream) {
  const float* feat = (const float*)d_in[0];
  const float* refs = (const float*)d_in[1];
  const float* sim  = (const float*)d_in[2];
  const int*   fidx = (const int*)d_in[3];
  const int*   pidx = (const int*)d_in[4];
  float* out = (float*)d_out;

  char* ws = (char*)d_ws;
  unsigned short* Qn  = (unsigned short*)ws;                                   // 4 MB
  unsigned short* RnT = (unsigned short*)(ws + (size_t)B_ * NUM_ * C_ * 2);    // 56 MB
  int* refIdx = (int*)(ws + (size_t)B_ * NUM_ * C_ * 2
                          + (size_t)B_ * NREF_ * HW_ * C_ * 2);                // 0.45 MB

  int n4 = B_ * C_ * HW_ / 4;
  hipLaunchKernelGGL(k_copy, dim3((n4 + 255) / 256), dim3(256), 0, stream,
                     (const float4*)feat, (float4*)out, n4);
  hipLaunchKernelGGL(k_qnorm, dim3(NUM_ / 256, B_), dim3(256), 0, stream, feat, fidx, Qn);
  hipLaunchKernelGGL(k_rnorm, dim3(HW_ / 256, B_ * NREF_), dim3(256), 0, stream, refs, pidx, RnT);
  hipLaunchKernelGGL(k_gemm_argmax, dim3(NUM_ / 128, B_ * NREF_), dim3(256), 0, stream,
                     Qn, RnT, refIdx);
  hipLaunchKernelGGL(k_fuse, dim3(B_ * NUM_), dim3(C_), 0, stream,
                     feat, refs, sim, fidx, refIdx, pidx, out);
  int ntot = B_ * NUM_ + B_ * NREF_ * NUM_;
  hipLaunchKernelGGL(k_idxout, dim3((ntot + 255) / 256), dim3(256), 0, stream,
                     fidx, refIdx, out);
}
